// GraphAttentionLayer_3951369912752
// MI455X (gfx1250) — matrix operand order repaired and checked by reading, not compiled, not run on hardware
//
#include <hip/hip_runtime.h>
#include <hip/hip_bf16.h>

// ---------------------------------------------------------------------------
// Fused graph-attention transformer layer for MI455X (gfx1250, wave32).
//
// Roofline: 95 GFLOP vs 350 MB mandatory HBM traffic (~15us @23.3TB/s); f16
// WMMA rate ~5 PFLOPS dense => GEMMs are critical path, so minimize padding
// and weight re-reads:
//   * pre-pass converts all weights once to f16, transposed [N][K], in d_ws
//   * 3 batch items per workgroup (M=64, 63 real rows -> 1.6% pad waste)
//   * B fragments read straight from global (two b128/lane, WGP$-cached),
//     A fragments from LDS; no weight staging, no per-chunk barriers
//   * tiny 21x21 masked attention in f32 VALU (4% of FLOPs)
// ---------------------------------------------------------------------------

typedef __attribute__((ext_vector_type(16))) _Float16 v16h;
typedef __attribute__((ext_vector_type(8)))  _Float16 v8h;
typedef __attribute__((ext_vector_type(8)))  float    v8f;

#define NJ    21
#define DTOK  128
#define FFD   256
#define GB    3          // batch items per workgroup
#define MR    64         // padded row count (3*21 = 63 real)
#define PA    136        // padded row pitch (halves) for 128-wide buffers
#define PF    264        // padded row pitch (halves) for 256-wide FF buffer

// ws (f16) layout offsets, all transposed [N][K]
#define OFF_WQ 0
#define OFF_WK 16384
#define OFF_WV 32768
#define OFF_WO 49152
#define OFF_W1 65536     // [256][128]
#define OFF_W2 98304     // [128][256]
#define WS_HALVES 131072 // 256 KB total

// adjacency (skeleton edges + self): bit j of ADJ[q] => key j visible to q
__device__ __constant__ unsigned ADJ[NJ] = {
    0x22223u, 0x7u,     0xEu,     0x1Cu,    0x18u,
    0x61u,    0xE0u,    0x1C0u,   0x180u,
    0x601u,   0xE00u,   0x1C00u,  0x1800u,
    0x6001u,  0xE000u,  0x1C000u, 0x18000u,
    0x60001u, 0xE0000u, 0x1C0000u,0x180000u
};

// ---------------------------------------------------------------------------
// Pre-pass: out[n*K + k] = (f16) W[k*N + n]   (transpose + f32->f16)
// ---------------------------------------------------------------------------
__global__ __launch_bounds__(256)
void convert_weights_kernel(const float* __restrict__ W, _Float16* __restrict__ out,
                            int K, int N)
{
    int i = blockIdx.x * 256 + threadIdx.x;
    if (i >= K * N) return;
    int k = i % K;
    int n = i / K;
    out[(size_t)n * K + k] = (_Float16)W[(size_t)k * N + n];
}

// ---------------------------------------------------------------------------
// GEMM stage: C[MR, NOUT] = A[MR, KTOT] (f16 LDS, pitch PA_) x W^T (f16 global
// [NOUT][KTOT]) + bias (f32 global), optional ReLU, -> O (f16 LDS, pitch PO_).
// 8 waves: wave w -> mtile (w&3), ntiles [(w>>2)*NPW, +NPW).
// ---------------------------------------------------------------------------
template<int KTOT, int NOUT, bool RELU>
__device__ inline void gemm_stage(const _Float16* __restrict__ A, int PA_,
                                  const _Float16* __restrict__ Wt,
                                  const float* __restrict__ bias,
                                  _Float16* __restrict__ O, int PO_,
                                  int lane, int wave)
{
    constexpr int NT  = NOUT / 16;
    constexpr int NPW = NT / 2;          // ntiles per wave (2 waves per mtile)

    const int mtile = wave & 3;
    const int nb    = (wave >> 2) * NPW; // first ntile of this wave
    const int lhalf = lane >> 4;
    const int lrow  = lane & 15;

    v8f acc[NPW];
    const v8f vzero = {0.f,0.f,0.f,0.f,0.f,0.f,0.f,0.f};
#pragma unroll
    for (int t = 0; t < NPW; ++t) acc[t] = vzero;

    const int arow = mtile * 16 + lrow;

    for (int kb = 0; kb < KTOT; kb += 32) {
        // A fragment (16x32 f16), hoisted across this wave's ntiles
        const int c0 = kb + lhalf * 8;
        v8h x0 = *(const v8h*)(A + arow * PA_ + c0);
        v8h x1 = *(const v8h*)(A + arow * PA_ + c0 + 16);
        v16h af;
#pragma unroll
        for (int i = 0; i < 8; ++i) { af[i] = x0[i]; af[8 + i] = x1[i]; }

        if (kb + 32 < KTOT)   // prefetch next weight chunk (global_prefetch_b8)
            __builtin_prefetch(Wt + (size_t)(nb * 16 + lrow) * KTOT + kb + 32, 0, 1);

#pragma unroll
        for (int t = 0; t < NPW; ++t) {
            const int n = (nb + t) * 16 + lrow;
            const v8h* bp = (const v8h*)(Wt + (size_t)n * KTOT + kb + lhalf * 8);
            v8h y0 = bp[0];
            v8h y1 = bp[2];
            v16h bf;
#pragma unroll
            for (int i = 0; i < 8; ++i) { bf[i] = y0[i]; bf[8 + i] = y1[i]; }

            acc[t] = __builtin_amdgcn_wmma_f32_16x16x32_f16(
                         false, af, false, bf, (short)0, acc[t], false, false);
        }
    }

    // epilogue: +bias (+ReLU), f16 store. C layout (ISA 7.12.2): lanes 0-15 ->
    // M = r, lanes 16-31 -> M = 8+r, N = lane&15.
    const int row0 = mtile * 16 + lhalf * 8;
#pragma unroll
    for (int t = 0; t < NPW; ++t) {
        const int n = (nb + t) * 16 + lrow;
        const float bv = bias[n];
#pragma unroll
        for (int r = 0; r < 8; ++r) {
            float f = acc[t][r] + bv;
            if (RELU) f = f > 0.f ? f : 0.f;
            O[(row0 + r) * PO_ + n] = (_Float16)f;
        }
    }
    __syncthreads();
}

// ---------------------------------------------------------------------------
__global__ __launch_bounds__(256)
void hand_attn_layer_kernel(const float* __restrict__ tokens,
                            const _Float16* __restrict__ wsh,   // converted weights
                            const float* __restrict__ bq, const float* __restrict__ bk,
                            const float* __restrict__ bv, const float* __restrict__ bo,
                            const float* __restrict__ g1, const float* __restrict__ beta1,
                            const float* __restrict__ g2, const float* __restrict__ beta2,
                            const float* __restrict__ bf1, const float* __restrict__ bf2,
                            float* __restrict__ out, int Btot)
{
    // LDS (halves): X[64][PA] tokens->ff2out, XL[64][PA] k->x, QB[64][PA]
    // q->attended, VB[64][PA] v->wo_out; FF[64][PF] aliased over QB+VB.
    __shared__ __align__(16) _Float16 S[4 * MR * PA];
    __shared__ float mu[MR], rs[MR];

    _Float16* X  = S;
    _Float16* XL = S + 1 * MR * PA;
    _Float16* QB = S + 2 * MR * PA;
    _Float16* VB = S + 3 * MR * PA;
    _Float16* FF = QB;                  // MR*PF = 16896 <= 2*MR*PA = 17408

    const int base = blockIdx.x * GB;   // first batch item of this WG
    const int tid  = threadIdx.x;
    const int lane = tid & 31;
    const int wave = tid >> 5;

    // ---- load GB items' tokens -> f16 LDS; pad rows & missing items -> 0 ---
    for (int idx = tid; idx < MR * DTOK; idx += 256) {
        int j = idx >> 7, d = idx & 127;
        int it = j / NJ, jj = j - it * NJ;
        float v = 0.f;
        if (j < GB * NJ && base + it < Btot)
            v = tokens[((size_t)(base + it) * NJ + jj) * DTOK + d];
        X[j * PA + d] = (_Float16)v;
    }
    __syncthreads();

    // ---- QKV projections ---------------------------------------------------
    gemm_stage<128, 128, false>(X, PA, wsh + OFF_WQ, bq, QB, PA, lane, wave);
    gemm_stage<128, 128, false>(X, PA, wsh + OFF_WK, bk, XL, PA, lane, wave);
    gemm_stage<128, 128, false>(X, PA, wsh + OFF_WV, bv, VB, PA, lane, wave);

    // ---- masked attention: head = wave, item = p, query row = lane ---------
    {
        const int h = wave;
        for (int p = 0; p < GB; ++p) {
            const int r0 = p * NJ;      // first local row of item p
            float qr[16];
#pragma unroll
            for (int d = 0; d < 16; ++d)
                qr[d] = (lane < NJ) ? (float)QB[(r0 + lane) * PA + h * 16 + d] : 0.f;

            const unsigned m = (lane < NJ) ? ADJ[lane] : 0u;
            float s[NJ];
            float mx = -1e30f;
#pragma unroll
            for (int j = 0; j < NJ; ++j) {
                float dot = 0.f;
#pragma unroll
                for (int d = 0; d < 16; ++d)
                    dot += qr[d] * (float)XL[(r0 + j) * PA + h * 16 + d];  // K
                float sc = dot * 0.25f;                                    // 1/sqrt(16)
                bool on = (m >> j) & 1u;
                s[j] = on ? sc : -1e30f;
                mx = (on && sc > mx) ? sc : mx;
            }
            float denom = 0.f;
#pragma unroll
            for (int j = 0; j < NJ; ++j) {
                float e = ((m >> j) & 1u) ? __expf(s[j] - mx) : 0.f;
                s[j] = e; denom += e;
            }
            const float inv = (denom > 0.f) ? 1.f / denom : 0.f;

            float acc[16];
#pragma unroll
            for (int d = 0; d < 16; ++d) acc[d] = 0.f;
#pragma unroll
            for (int j = 0; j < NJ; ++j) {
                float a = s[j];
#pragma unroll
                for (int d = 0; d < 16; ++d)
                    acc[d] += a * (float)VB[(r0 + j) * PA + h * 16 + d];
            }
            if (lane < NJ) {            // overwrite q with attended (head cols)
#pragma unroll
                for (int d = 0; d < 16; ++d)
                    QB[(r0 + lane) * PA + h * 16 + d] = (_Float16)(acc[d] * inv);
            }
        }
    }
    __syncthreads();

    // ---- output projection: attended @ Wo -> VB ----------------------------
    gemm_stage<128, 128, false>(QB, PA, wsh + OFF_WO, bo, VB, PA, lane, wave);

    // ---- LN1: x = LN(tokens + wo_out) -> XL --------------------------------
    if (tid < MR) {
        const int j = tid;
        float m1 = 0.f;
        for (int d = 0; d < DTOK; ++d)
            m1 += (float)X[j * PA + d] + (float)VB[j * PA + d];
        m1 *= (1.f / 128.f);
        float v1 = 0.f;
        for (int d = 0; d < DTOK; ++d) {
            float t = (float)X[j * PA + d] + (float)VB[j * PA + d] - m1;
            v1 += t * t;
        }
        mu[j] = m1;
        rs[j] = rsqrtf(v1 * (1.f / 128.f) + 1e-5f);
    }
    __syncthreads();
    for (int idx = tid; idx < MR * DTOK; idx += 256) {
        int j = idx >> 7, d = idx & 127;
        float val = 0.f;
        if (j < GB * NJ)
            val = (((float)X[j * PA + d] + (float)VB[j * PA + d]) - mu[j])
                  * rs[j] * g1[d] + beta1[d];
        XL[j * PA + d] = (_Float16)val;     // pad rows -> 0
    }
    __syncthreads();

    // ---- FF1: relu(x @ W1 + bf1) -> FF (aliases QB+VB) ---------------------
    gemm_stage<128, 256, true >(XL, PA, wsh + OFF_W1, bf1, FF, PF, lane, wave);

    // ---- FF2: ff @ W2 + bf2 -> X (tokens dead) -----------------------------
    gemm_stage<256, 128, false>(FF, PF, wsh + OFF_W2, bf2, X,  PA, lane, wave);

    // ---- LN2: out = LN(x + ff2) -> global ----------------------------------
    if (tid < MR) {
        const int j = tid;
        float m1 = 0.f;
        for (int d = 0; d < DTOK; ++d)
            m1 += (float)X[j * PA + d] + (float)XL[j * PA + d];
        m1 *= (1.f / 128.f);
        float v1 = 0.f;
        for (int d = 0; d < DTOK; ++d) {
            float t = (float)X[j * PA + d] + (float)XL[j * PA + d] - m1;
            v1 += t * t;
        }
        mu[j] = m1;
        rs[j] = rsqrtf(v1 * (1.f / 128.f) + 1e-5f);
    }
    __syncthreads();

    const int nv = (Btot - base < GB) ? (Btot - base) : GB;   // valid items
    float* orow = out + (size_t)base * NJ * DTOK;
    for (int idx = tid; idx < nv * NJ * DTOK; idx += 256) {
        int j = idx >> 7, d = idx & 127;
        float val = (((float)X[j * PA + d] + (float)XL[j * PA + d]) - mu[j])
                    * rs[j] * g2[d] + beta2[d];
        orow[idx] = val;
    }
}

// ---------------------------------------------------------------------------
extern "C" void kernel_launch(void* const* d_in, const int* in_sizes, int n_in,
                              void* d_out, int out_size, void* d_ws, size_t ws_size,
                              hipStream_t stream)
{
    const float* tokens = (const float*)d_in[0];
    const float* Wq     = (const float*)d_in[1];
    const float* bq     = (const float*)d_in[2];
    const float* Wk     = (const float*)d_in[3];
    const float* bk     = (const float*)d_in[4];
    const float* Wv     = (const float*)d_in[5];
    const float* bv     = (const float*)d_in[6];
    const float* Wo     = (const float*)d_in[7];
    const float* bo     = (const float*)d_in[8];
    const float* g1     = (const float*)d_in[9];
    const float* beta1  = (const float*)d_in[10];
    const float* g2     = (const float*)d_in[11];
    const float* beta2  = (const float*)d_in[12];
    const float* W1     = (const float*)d_in[13];
    const float* bf1    = (const float*)d_in[14];
    const float* W2     = (const float*)d_in[15];
    const float* bf2    = (const float*)d_in[16];

    _Float16* wsh = (_Float16*)d_ws;    // needs >= 256 KB (WS_HALVES * 2 B)

    // pre-pass: f32 -> f16 transposed weight images (one-time, 512 KB read)
    convert_weights_kernel<<<dim3((128*128 + 255) / 256), dim3(256), 0, stream>>>(
        Wq, wsh + OFF_WQ, 128, 128);
    convert_weights_kernel<<<dim3((128*128 + 255) / 256), dim3(256), 0, stream>>>(
        Wk, wsh + OFF_WK, 128, 128);
    convert_weights_kernel<<<dim3((128*128 + 255) / 256), dim3(256), 0, stream>>>(
        Wv, wsh + OFF_WV, 128, 128);
    convert_weights_kernel<<<dim3((128*128 + 255) / 256), dim3(256), 0, stream>>>(
        Wo, wsh + OFF_WO, 128, 128);
    convert_weights_kernel<<<dim3((128*256 + 255) / 256), dim3(256), 0, stream>>>(
        W1, wsh + OFF_W1, 128, 256);   // -> [256][128]
    convert_weights_kernel<<<dim3((256*128 + 255) / 256), dim3(256), 0, stream>>>(
        W2, wsh + OFF_W2, 256, 128);   // -> [128][256]

    const int Btot   = in_sizes[0] / (NJ * DTOK);   // 16384
    const int blocks = (Btot + GB - 1) / GB;

    hand_attn_layer_kernel<<<dim3(blocks), dim3(256), 0, stream>>>(
        tokens, wsh, bq, bk, bv, bo, g1, beta1, g2, beta2, bf1, bf2,
        (float*)d_out, Btot);

    (void)ws_size; (void)n_in; (void)out_size;
}
